// FluteLinear_49658411876686
// MI455X (gfx1250) — compile-verified, matrix operand order checked by reading, not run on hardware
//
#include <hip/hip_runtime.h>

// CDNA5 WMMA vector types
typedef __attribute__((ext_vector_type(16))) _Float16 v16h;
typedef __attribute__((ext_vector_type(8)))  _Float16 v8h;
typedef __attribute__((ext_vector_type(8)))  float    v8f;

#define N_DIM   14336
#define K_DIM   4096
#define G_DIM   32            // K / GROUP_SIZE
#define GS      128           // group size along K
#define M_DIM   16            // BATCH
#define WAVES   8             // waves per block (wave32 -> 256 threads)
#define K_PER_WAVE (K_DIM / WAVES)     // 512
#define GROUPS_PER_WAVE (K_PER_WAVE / GS)  // 4
#define ITERS_PER_GROUP (GS / 32)          // 4 WMMA steps per group

// ---------------------------------------------------------------------------
// Pre-pass: convert x[16,4096] f32 -> f16 once (it was re-converted 896x
// before). 64K elements, trivial cost, removes all A-side cvt from main loop.
// ---------------------------------------------------------------------------
typedef __attribute__((ext_vector_type(4))) _Float16 v4h;

__global__ __launch_bounds__(256) void cvt_x_kernel(
    const float* __restrict__ x, _Float16* __restrict__ x16)
{
    const int i = (blockIdx.x * 256 + threadIdx.x) * 4;  // 4 elems / thread
    const float4 v = *(const float4*)(x + i);
    v4h h;
    h[0] = (_Float16)v.x; h[1] = (_Float16)v.y;
    h[2] = (_Float16)v.z; h[3] = (_Float16)v.w;
    *(v4h*)(x16 + i) = h;
}

// ---------------------------------------------------------------------------
// Main kernel: one block = one 16-wide output tile; 8 waves split K.
// B dequant: 256-entry LDS pair-table of packed f16 (t[lo], t[hi]);
// group scale folded into the f32 accumulator once per 128-wide K group.
// ---------------------------------------------------------------------------
__global__ __launch_bounds__(256) void flute_wmma_kernel(
    const _Float16* __restrict__ x16,     // [16, 4096] f16 (from d_ws)
    const int*      __restrict__ qw,      // [14336, 4096] int32 codes 0..15
    const float*    __restrict__ scales,  // [14336, 32] f32
    const float*    __restrict__ tables,  // [16] f32
    float*          __restrict__ out)     // [16, 14336] f32
{
    __shared__ unsigned s_pair[256];      // packed f16 pair per (lo,hi) code pair
    __shared__ float    s_red[WAVES * 256];

    const int tid = threadIdx.x;
    {
        // pair j -> (f16)tables[j&15] in low half, (f16)tables[j>>4] in high half
        const _Float16 lo = (_Float16)tables[tid & 15];
        const _Float16 hi = (_Float16)tables[tid >> 4];
        const unsigned ulo = (unsigned)__builtin_bit_cast(unsigned short, lo);
        const unsigned uhi = (unsigned)__builtin_bit_cast(unsigned short, hi);
        s_pair[tid] = ulo | (uhi << 16);
    }
    __syncthreads();

    const int wave = tid >> 5;
    const int lane = tid & 31;
    const int half = lane >> 4;   // 0: K offsets {0..7,16..23}; 1: {8..15,24..31}
    const int r    = lane & 15;   // A row (m) / B column offset (n)

    const int n0 = blockIdx.x * 16;
    const int n  = n0 + r;

    const _Float16* xrow = x16 + (size_t)r * K_DIM;
    const int*      qrow = qw  + (size_t)n * K_DIM;
    const float*    srow = scales + (size_t)n * G_DIM;

    v8f c = {};

    const int g0 = wave * GROUPS_PER_WAVE;

    for (int g = 0; g < GROUPS_PER_WAVE; ++g) {
        const float s = srow[g0 + g];     // one scalar load per 128-wide group
        v8f cg = {};                      // unscaled group accumulator

        #pragma unroll
        for (int it = 0; it < ITERS_PER_GROUP; ++it) {
            const int k0 = wave * K_PER_WAVE + g * GS + it * 32;
            const int kb = k0 + half * 8;

            // keep the streaming pipe fed (global_prefetch_b8)
            __builtin_prefetch(qrow + kb + 32, 0, 0);

            // ---- A tile: already f16 in WMMA-friendly row-major layout ----
            union { v16h h; uint4 u[2]; } aa;
            aa.u[0] = *(const uint4*)(xrow + kb);        // K = kb .. kb+7
            aa.u[1] = *(const uint4*)(xrow + kb + 16);   // K = kb+16 .. kb+23

            // ---- B tile: 16 codes -> 8 packed-pair LDS lookups ----
            const int4 q0 = *(const int4*)(qrow + kb);
            const int4 q1 = *(const int4*)(qrow + kb + 4);
            const int4 q2 = *(const int4*)(qrow + kb + 16);
            const int4 q3 = *(const int4*)(qrow + kb + 20);

            union { v16h h; unsigned u[8]; } bb;
            bb.u[0] = s_pair[q0.x + (q0.y << 4)];
            bb.u[1] = s_pair[q0.z + (q0.w << 4)];
            bb.u[2] = s_pair[q1.x + (q1.y << 4)];
            bb.u[3] = s_pair[q1.z + (q1.w << 4)];
            bb.u[4] = s_pair[q2.x + (q2.y << 4)];
            bb.u[5] = s_pair[q2.z + (q2.w << 4)];
            bb.u[6] = s_pair[q3.x + (q3.y << 4)];
            bb.u[7] = s_pair[q3.z + (q3.w << 4)];

            // D = A(16x32) * B(32x16) + C  -> v_wmma_f32_16x16x32_f16
            cg = __builtin_amdgcn_wmma_f32_16x16x32_f16(
                    /*neg_a=*/false, aa.h, /*neg_b=*/false, bb.h,
                    /*c_mod=*/(short)0, cg, /*reuse_a=*/false, /*reuse_b=*/false);
        }

        // fold group scale: c += s * cg   (lane's N index is constant -> legal)
        #pragma unroll
        for (int j = 0; j < 8; ++j) c[j] += s * cg[j];
    }

    // ---- cross-wave reduction of the 16x16 f32 tile via LDS ----
    {
        float* dst = s_red + wave * 256 + lane * 8;
        #pragma unroll
        for (int j = 0; j < 8; ++j) dst[j] = c[j];
    }
    __syncthreads();

    {
        // 256 threads cover the 256 tile elements; element i was produced at
        // lane le = i/8, accumulator VGPR j = i%8.
        const int i  = tid;
        const int le = i >> 3;
        const int j  = i & 7;
        float acc = 0.0f;
        #pragma unroll
        for (int w = 0; w < WAVES; ++w) acc += s_red[w * 256 + i];
        // C/D VGPR layout: lanes 0-15 -> M=j, N=lane; lanes 16-31 -> M=j+8, N=lane-16
        const int m  = j + ((le >> 4) << 3);
        const int nn = n0 + (le & 15);
        out[(size_t)m * N_DIM + nn] = acc;
    }
}

extern "C" void kernel_launch(void* const* d_in, const int* in_sizes, int n_in,
                              void* d_out, int out_size, void* d_ws, size_t ws_size,
                              hipStream_t stream) {
    const float* x      = (const float*)d_in[0];
    const int*   qw     = (const int*)d_in[1];
    const float* scales = (const float*)d_in[2];
    const float* tables = (const float*)d_in[3];
    float*       out    = (float*)d_out;

    _Float16* x16 = (_Float16*)d_ws;   // 16*4096*2B = 128 KB scratch

    // Pre-pass: x f32 -> f16 (64K elements, 4 per thread)
    cvt_x_kernel<<<dim3((M_DIM * K_DIM) / (256 * 4)), dim3(256), 0, stream>>>(x, x16);

    // Main GEMM: 896 blocks x 256 threads (8 wave32 waves, K split 8 ways)
    flute_wmma_kernel<<<dim3(N_DIM / 16), dim3(256), 0, stream>>>(
        x16, qw, scales, tables, out);
}
